// MultiAttentionHead_35055523070492
// MI455X (gfx1250) — matrix operand order repaired and checked by reading, not compiled, not run on hardware
//
#include <hip/hip_runtime.h>
#include <math.h>

// ---------------------------------------------------------------------------
// MI455X (gfx1250) multi-head causal attention, bf16 WMMA pipeline.
// B=2, S=2048, D=1024, H=16, DK=64.  All matmuls on v_wmma_f32_16x16x32_bf16.
// Round 2: software-pipelined fragment loads (ping-pong register sets) so the
// wmma bursts are not serialized behind per-fragment s_wait_loadcnt 0.
// ---------------------------------------------------------------------------

typedef __bf16 bf16;
typedef __attribute__((ext_vector_type(16))) __bf16 v16bf;
typedef __attribute__((ext_vector_type(8)))  __bf16 v8bf;   // 16 bytes -> b128
typedef __attribute__((ext_vector_type(8)))  float  v8f;

#define B_  2
#define S_  2048
#define D_  1024
#define H_  16
#define DK_ 64
#define M_  (B_ * S_)          // 4096 token rows
#define NEG_INF (-__builtin_huge_valf())

union FragU { v16bf v; v8bf h[2]; };

// Build a 16-element bf16 fragment from two contiguous 16-byte halves.
__device__ __forceinline__ v16bf load_frag(const bf16* p0, const bf16* p1) {
  FragU u;
  u.h[0] = *reinterpret_cast<const v8bf*>(p0);
  u.h[1] = *reinterpret_cast<const v8bf*>(p1);
  return u.v;
}

__device__ __forceinline__ v8f wmma_bf16(v16bf a, v16bf b, v8f c) {
  // D = A(16x32 bf16) * B(32x16 bf16) + C(16x16 f32)
  return __builtin_amdgcn_wmma_f32_16x16x32_bf16(false, a, false, b,
                                                 (short)0, c, false, false);
}

__device__ __forceinline__ v8f v8f_zero() {
  v8f z = {0.f, 0.f, 0.f, 0.f, 0.f, 0.f, 0.f, 0.f};
  return z;
}

// Row reductions across the 16-lane half-wave (wave32; C-tile rows span
// lanes 0-15 / 16-31 independently, xor masks <= 8 stay inside each half).
__device__ __forceinline__ float rowmax16(float v) {
  v = fmaxf(v, __shfl_xor(v, 1, 32));
  v = fmaxf(v, __shfl_xor(v, 2, 32));
  v = fmaxf(v, __shfl_xor(v, 4, 32));
  v = fmaxf(v, __shfl_xor(v, 8, 32));
  return v;
}
__device__ __forceinline__ float rowsum16(float v) {
  v += __shfl_xor(v, 1, 32);
  v += __shfl_xor(v, 2, 32);
  v += __shfl_xor(v, 4, 32);
  v += __shfl_xor(v, 8, 32);
  return v;
}

// ---------------------------------------------------------------------------
// Kernel 1: x -> bf16 (row-major), W -> bf16 transposed (W^T[n][k]) so GEMM
// B-fragments are two contiguous 16B loads per lane.
// ---------------------------------------------------------------------------
__global__ void cvt_kernel(const float* __restrict__ x,
                           const float* __restrict__ Wq, const float* __restrict__ Wk,
                           const float* __restrict__ Wv, const float* __restrict__ Wo,
                           bf16* __restrict__ xb,
                           bf16* __restrict__ WqT, bf16* __restrict__ WkT,
                           bf16* __restrict__ WvT, bf16* __restrict__ WoT) {
  int i = blockIdx.x * 256 + threadIdx.x;              // 0 .. 4M-1
  xb[i] = (bf16)x[i];
  if (i < D_ * D_) {
    int k = i >> 10, n = i & (D_ - 1);
    size_t t = (size_t)n * D_ + k;                     // W^T[n][k] = W[k][n]
    WqT[t] = (bf16)Wq[i];
    WkT[t] = (bf16)Wk[i];
    WvT[t] = (bf16)Wv[i];
    WoT[t] = (bf16)Wo[i];
  }
}

// ---------------------------------------------------------------------------
// Kernel 2: fused QKV projection.  Wave tile = 16 rows x 64 cols, 3 weights.
// Q,K stored [B*H][S][DK]; V stored transposed [B*H][DK][S].
// B-fragments double-buffered: group0 = n-subtiles {0,1}, group1 = {2,3}.
// ---------------------------------------------------------------------------
__global__ void __launch_bounds__(256)
qkv_gemm(const bf16* __restrict__ Xb,
         const bf16* __restrict__ WqT, const bf16* __restrict__ WkT,
         const bf16* __restrict__ WvT,
         bf16* __restrict__ Q, bf16* __restrict__ K, bf16* __restrict__ Vt) {
  const int lane = threadIdx.x & 31;
  const int w    = threadIdx.x >> 5;
  const int tile = blockIdx.x * 8 + w;       // 4096 wave tiles
  const int mt = tile >> 4, nt = tile & 15;
  const int mb = mt * 16, nb = nt * 64;
  const int lo = lane & 15, hi = lane >> 4;
  const int kb = hi * 8;                     // A-fragment k base for this half

  const bf16* Wt[3] = {WqT, WkT, WvT};
  v8f acc[3][4];
#pragma unroll
  for (int s = 0; s < 3; ++s)
#pragma unroll
    for (int n4 = 0; n4 < 4; ++n4) acc[s][n4] = v8f_zero();

  auto loadA = [&](int k0) -> v16bf {
    const bf16* xs = Xb + (size_t)(mb + lo) * D_ + k0;
    return load_frag(xs + kb, xs + 16 + kb);           // A: 16x32 bf16
  };
  auto loadB = [&](v16bf (&dst)[2][3], int n4base, int k0) {
#pragma unroll
    for (int n4 = 0; n4 < 2; ++n4)
#pragma unroll
      for (int s = 0; s < 3; ++s) {
        const bf16* ws = Wt[s] + (size_t)(nb + (n4base + n4) * 16 + lo) * D_
                       + k0 + hi * 8;
        dst[n4][s] = load_frag(ws, ws + 16);           // B: 32x16 bf16
      }
  };

  v16bf bA[2][3], bB[2][3];
  v16bf a = loadA(0);
  loadB(bA, 0, 0);                                     // group0 of k=0
  for (int k0 = 0; k0 < D_; k0 += 32) {
    loadB(bB, 2, k0);                                  // group1 of k0 in flight
#pragma unroll
    for (int n4 = 0; n4 < 2; ++n4)
#pragma unroll
      for (int s = 0; s < 3; ++s)
        acc[s][n4] = wmma_bf16(a, bA[n4][s], acc[s][n4]);
    const int k1 = (k0 + 32 < D_) ? k0 + 32 : 0;       // dummy reload on last
    v16bf an = loadA(k1);
    loadB(bA, 0, k1);                                  // group0 of next k
#pragma unroll
    for (int n4 = 0; n4 < 2; ++n4)
#pragma unroll
      for (int s = 0; s < 3; ++s)
        acc[s][n4 + 2] = wmma_bf16(a, bB[n4][s], acc[s][n4 + 2]);
    a = an;
  }
  // Scatter-store C tiles (lane holds col nb+n4*16+lo, VGPR r -> row r+hi*8)
#pragma unroll
  for (int r = 0; r < 8; ++r) {
    const int mrow = mb + r + hi * 8;
    const int b = mrow >> 11, s = mrow & (S_ - 1);
#pragma unroll
    for (int n4 = 0; n4 < 4; ++n4) {
      const int n = nb + n4 * 16 + lo;
      const int h = n >> 6, dk = n & 63;
      const size_t bh = (size_t)(b * H_ + h);
      Q [(bh * S_ + s) * DK_ + dk] = (bf16)acc[0][n4][r];
      K [(bh * S_ + s) * DK_ + dk] = (bf16)acc[1][n4][r];
      Vt[(bh * DK_ + dk) * S_ + s] = (bf16)acc[2][n4][r];
    }
  }
}

// ---------------------------------------------------------------------------
// Kernel 3: flash attention.  One wave per 16 q-rows; 32-key chunks with
// online softmax; P relayout (C-tile -> A-fragment) through per-wave LDS.
// K-fragments batched before the score wmmas; V-fragments hoisted above the
// softmax VALU section so their latency hides under exp/shuffle math.
// ---------------------------------------------------------------------------
__global__ void __launch_bounds__(256)
flash_attn(const bf16* __restrict__ Q, const bf16* __restrict__ K,
           const bf16* __restrict__ Vt, bf16* __restrict__ O) {
  const int lane = threadIdx.x & 31;
  const int w    = threadIdx.x >> 5;
  const int bh   = blockIdx.x >> 4;               // 0..31 (b*H+h)
  const int qc   = blockIdx.x & 15;
  const int qb   = qc * 128 + w * 16;             // this wave's q-row base
  const int b    = bh >> 4, h = bh & 15;
  const int lo = lane & 15, hi = lane >> 4;
  const int kb = hi * 8;

  const bf16* Qb = Q  + (size_t)bh * S_ * DK_;
  const bf16* Kb = K  + (size_t)bh * S_ * DK_;
  const bf16* Vb = Vt + (size_t)bh * DK_ * S_;

  // Q A-fragments: frag f covers dk [32f, 32f+32)
  v16bf qa[2];
#pragma unroll
  for (int f = 0; f < 2; ++f) {
    const bf16* qs = Qb + (size_t)(qb + lo) * DK_ + f * 32;
    qa[f] = load_frag(qs + kb, qs + 16 + kb);
  }

  v8f o[4];
#pragma unroll
  for (int d = 0; d < 4; ++d) o[d] = v8f_zero();
  float m_i[8], l_i[8];
#pragma unroll
  for (int r = 0; r < 8; ++r) { m_i[r] = NEG_INF; l_i[r] = 0.f; }

  __shared__ __align__(16) bf16 pbuf[8][16][32];  // 8 KB, one slab per wave
  bf16 (*pw)[32] = pbuf[w];

  const int kend = qb + 16;                        // causal: keys <= q row
  for (int j = 0; j < kend; j += 32) {
    // ---- batched K fragments, then score wmma burst ----
    v16bf kf[2][2];                                // [tile t][dk chunk f]
#pragma unroll
    for (int t = 0; t < 2; ++t)
#pragma unroll
      for (int f = 0; f < 2; ++f) {
        const bf16* ks = Kb + (size_t)(j + t * 16 + lo) * DK_ + f * 32 + hi * 8;
        kf[t][f] = load_frag(ks, ks + 16);
      }
    v8f sc0 = v8f_zero(), sc1 = v8f_zero();
    sc0 = wmma_bf16(qa[0], kf[0][0], sc0);
    sc0 = wmma_bf16(qa[1], kf[0][1], sc0);
    sc1 = wmma_bf16(qa[0], kf[1][0], sc1);
    sc1 = wmma_bf16(qa[1], kf[1][1], sc1);
    // ---- V fragments in flight while softmax VALU runs ----
    v16bf vf[4];
#pragma unroll
    for (int d = 0; d < 4; ++d) {
      const bf16* vs = Vb + (size_t)(d * 16 + lo) * S_ + j + hi * 8;
      vf[d] = load_frag(vs, vs + 16);
    }
    // ---- online softmax (per C-tile row r; row = qb + r + hi*8) ----
#pragma unroll
    for (int r = 0; r < 8; ++r) {
      const int qrow = qb + r + hi * 8;
      float s0 = sc0[r] * 0.125f;                  // 1/sqrt(DK)
      float s1 = sc1[r] * 0.125f;
      if (j + lo > qrow)      s0 = NEG_INF;        // causal mask
      if (j + 16 + lo > qrow) s1 = NEG_INF;
      const float mnew = fmaxf(m_i[r], rowmax16(fmaxf(s0, s1)));
      const float corr = __expf(m_i[r] - mnew);    // exp(-inf)=0 first pass
      const float p0 = __expf(s0 - mnew);
      const float p1 = __expf(s1 - mnew);
      l_i[r] = l_i[r] * corr + rowsum16(p0 + p1);
      m_i[r] = mnew;
#pragma unroll
      for (int d = 0; d < 4; ++d) o[d][r] *= corr; // rescale accumulator
      const int row = r + hi * 8;
      pw[row][lo]      = (bf16)p0;                 // stage P tile in LDS
      pw[row][16 + lo] = (bf16)p1;
    }
    asm volatile("s_wait_dscnt 0" ::: "memory");   // in-wave LDS RAW fence
    // ---- P (16x32) as A-fragment, accumulate P @ V ----
    const bf16* ps = &pw[lo][0];
    v16bf pa = load_frag(ps + kb, ps + 16 + kb);
#pragma unroll
    for (int d = 0; d < 4; ++d)
      o[d] = wmma_bf16(pa, vf[d], o[d]);
  }
  // ---- finalize: O /= l, store bf16 into [B*S][D] head-concat layout ----
#pragma unroll
  for (int r = 0; r < 8; ++r) {
    const float inv = 1.0f / l_i[r];
    const size_t tok = (size_t)b * S_ + qb + r + hi * 8;
#pragma unroll
    for (int d = 0; d < 4; ++d)
      O[tok * D_ + h * 64 + d * 16 + lo] = (bf16)(o[d][r] * inv);
  }
}

// ---------------------------------------------------------------------------
// Kernel 4: output projection, fp32 result.  Same ping-pong pipelining.
// ---------------------------------------------------------------------------
__global__ void __launch_bounds__(256)
out_gemm(const bf16* __restrict__ Ob, const bf16* __restrict__ WoT,
         float* __restrict__ out) {
  const int lane = threadIdx.x & 31;
  const int w    = threadIdx.x >> 5;
  const int tile = blockIdx.x * 8 + w;
  const int mt = tile >> 4, nt = tile & 15;
  const int mb = mt * 16, nb = nt * 64;
  const int lo = lane & 15, hi = lane >> 4;
  const int kb = hi * 8;

  v8f acc[4];
#pragma unroll
  for (int n4 = 0; n4 < 4; ++n4) acc[n4] = v8f_zero();

  auto loadA = [&](int k0) -> v16bf {
    const bf16* xs = Ob + (size_t)(mb + lo) * D_ + k0;
    return load_frag(xs + kb, xs + 16 + kb);
  };
  auto loadB = [&](v16bf (&dst)[2], int n4base, int k0) {
#pragma unroll
    for (int n4 = 0; n4 < 2; ++n4) {
      const bf16* ws = WoT + (size_t)(nb + (n4base + n4) * 16 + lo) * D_
                     + k0 + hi * 8;
      dst[n4] = load_frag(ws, ws + 16);
    }
  };

  v16bf bA[2], bB[2];
  v16bf a = loadA(0);
  loadB(bA, 0, 0);
  for (int k0 = 0; k0 < D_; k0 += 32) {
    loadB(bB, 2, k0);
#pragma unroll
    for (int n4 = 0; n4 < 2; ++n4) acc[n4] = wmma_bf16(a, bA[n4], acc[n4]);
    const int k1 = (k0 + 32 < D_) ? k0 + 32 : 0;
    v16bf an = loadA(k1);
    loadB(bA, 0, k1);
#pragma unroll
    for (int n4 = 0; n4 < 2; ++n4) acc[n4 + 2] = wmma_bf16(a, bB[n4], acc[n4 + 2]);
    a = an;
  }
#pragma unroll
  for (int r = 0; r < 8; ++r) {
    const int mrow = mb + r + hi * 8;
#pragma unroll
    for (int n4 = 0; n4 < 4; ++n4)
      out[(size_t)mrow * D_ + nb + n4 * 16 + lo] = acc[n4][r];
  }
}

// ---------------------------------------------------------------------------
extern "C" void kernel_launch(void* const* d_in, const int* in_sizes, int n_in,
                              void* d_out, int out_size, void* d_ws, size_t ws_size,
                              hipStream_t stream) {
  const float* x  = (const float*)d_in[0];
  const float* Wq = (const float*)d_in[1];
  const float* Wk = (const float*)d_in[2];
  const float* Wv = (const float*)d_in[3];
  const float* Wo = (const float*)d_in[4];
  float* out = (float*)d_out;

  // Workspace layout (bf16 elements): 24M elems = 48 MB total.
  bf16* ws  = (bf16*)d_ws;
  bf16* xb  = ws;                                   // [4096][1024]
  bf16* WqT = xb  + (size_t)M_ * D_;                // [1024][1024] each
  bf16* WkT = WqT + (size_t)D_ * D_;
  bf16* WvT = WkT + (size_t)D_ * D_;
  bf16* WoT = WvT + (size_t)D_ * D_;
  bf16* Qb  = WoT + (size_t)D_ * D_;                // [B*H][S][64]
  bf16* Kb  = Qb  + (size_t)B_ * H_ * S_ * DK_;
  bf16* Vtb = Kb  + (size_t)B_ * H_ * S_ * DK_;     // [B*H][64][S]
  bf16* Ob  = Vtb + (size_t)B_ * H_ * S_ * DK_;     // [4096][1024]

  cvt_kernel<<<(M_ * D_) / 256, 256, 0, stream>>>(x, Wq, Wk, Wv, Wo,
                                                  xb, WqT, WkT, WvT, WoT);
  qkv_gemm<<<512, 256, 0, stream>>>(xb, WqT, WkT, WvT, Qb, Kb, Vtb);
  flash_attn<<<512, 256, 0, stream>>>(Qb, Kb, Vtb, Ob);
  out_gemm<<<512, 256, 0, stream>>>(Ob, WoT, out);
}
